// MultiHeadAttention_23862838297096
// MI455X (gfx1250) — compile-verified, matrix operand order checked by reading, not run on hardware
//
#include <hip/hip_runtime.h>
#include <stdint.h>

#define D_MODEL 1024
#define N_HEADS 16
#define BB 2
#define SS 2048
#define DK 64
#define MTOT (BB * SS)   // 4096 rows
#define KSTEPS (D_MODEL / 32)

typedef __bf16 bf16;
typedef bf16  v16bf __attribute__((ext_vector_type(16)));
typedef float v8f   __attribute__((ext_vector_type(8)));
typedef unsigned int v4u __attribute__((ext_vector_type(4)));
typedef int v4i __attribute__((ext_vector_type(4)));
typedef int v8i __attribute__((ext_vector_type(8)));

#if __has_builtin(__builtin_amdgcn_tensor_load_to_lds)
#define HAVE_TDM 1
#else
#define HAVE_TDM 0
#endif

// ---------- helpers ----------

static __device__ __forceinline__ uint16_t f2bf(float f) {
  uint32_t u = __float_as_uint(f);
  u += 0x7fffu + ((u >> 16) & 1u);     // round-to-nearest-even
  return (uint16_t)(u >> 16);
}

static __device__ __forceinline__ v8f v8f_zero() {
  v8f v;
#pragma unroll
  for (int i = 0; i < 8; i++) v[i] = 0.0f;
  return v;
}

// Load a 16x32 bf16 A/B fragment for v_wmma_f32_16x16x32_bf16.
// lane (row = lane&15, half = lane>>4): dwords 0-3 hold k = 8*half..+7,
// dwords 4-7 hold k = 16+8*half..+7 (two contiguous 16-byte loads).
static __device__ __forceinline__ v16bf load_frag(const uint16_t* base, int row,
                                                  int ld, int half) {
  const uint16_t* p = base + (size_t)row * ld + 8 * half;
  union { uint4 q[2]; v16bf v; } u;
  u.q[0] = *(const uint4*)(p);
  u.q[1] = *(const uint4*)(p + 16);
  return u.v;
}

static __device__ __forceinline__ v8f wmma_bf16f32(v16bf a, v16bf b, v8f c) {
  return __builtin_amdgcn_wmma_f32_16x16x32_bf16(false, a, false, b, (short)0, c,
                                                 false, false);
}

static __device__ __forceinline__ float rowmax16(float v) {
  v = fmaxf(v, __shfl_xor(v, 1, 32));
  v = fmaxf(v, __shfl_xor(v, 2, 32));
  v = fmaxf(v, __shfl_xor(v, 4, 32));
  v = fmaxf(v, __shfl_xor(v, 8, 32));
  return v;
}
static __device__ __forceinline__ float rowsum16(float v) {
  v += __shfl_xor(v, 1, 32);
  v += __shfl_xor(v, 2, 32);
  v += __shfl_xor(v, 4, 32);
  v += __shfl_xor(v, 8, 32);
  return v;
}

// ---------- Tensor Data Mover descriptor construction (D#) ----------

static __device__ __forceinline__ void desc_set(uint32_t* d, int lo, int width,
                                                uint64_t v) {
  uint64_t mask = (width >= 64) ? ~0ull : ((1ull << width) - 1ull);
  v &= mask;
  int idx = lo >> 5, sh = lo & 31;
  d[idx] |= (uint32_t)(v << sh);
  if (sh + width > 32) d[idx + 1] |= (uint32_t)(v >> (32 - sh));
}

// Group0: count=1, lds_addr[63:32], global_addr[120:64], type=2 [127:126]
static __device__ __forceinline__ v4u make_g0(uint64_t gaddr, uint32_t lds_off) {
  v4u r;
  r[0] = 1u;
  r[1] = lds_off;
  r[2] = (uint32_t)gaddr;
  r[3] = (uint32_t)((gaddr >> 32) & 0x1FFFFFFu) | (2u << 30);
  return r;
}

// Group1: data_size(2B)@16, tensor_dim0@48/32, tensor_dim1@80/32,
// tile_dim0@112/16, tile_dim1@128/16, tensor_dim0_stride@160/48
static __device__ __forceinline__ v8i make_g1(uint32_t dim0, uint32_t dim1,
                                              uint32_t tile0, uint32_t tile1,
                                              uint64_t stride0) {
  uint32_t d[8] = {0, 0, 0, 0, 0, 0, 0, 0};
  desc_set(d, 16, 2, 1);
  desc_set(d, 48, 32, dim0);
  desc_set(d, 80, 32, dim1);
  desc_set(d, 112, 16, tile0);
  desc_set(d, 128, 16, tile1);
  desc_set(d, 160, 48, stride0);
  v8i r;
#pragma unroll
  for (int i = 0; i < 8; i++) r[i] = (int)d[i];
  return r;
}

#if HAVE_TDM
static __device__ __forceinline__ void tdm_load(v4u g0, v8i g1) {
  v4i z4; z4[0] = 0; z4[1] = 0; z4[2] = 0; z4[3] = 0;
#if defined(__clang_major__) && (__clang_major__ >= 23)
  v8i z8;
#pragma unroll
  for (int i = 0; i < 8; i++) z8[i] = 0;
  __builtin_amdgcn_tensor_load_to_lds(g0, g1, z4, z4, z8, 0);
#else
  __builtin_amdgcn_tensor_load_to_lds(g0, g1, z4, z4, 0);
#endif
}
#endif

// ---------- kernels ----------

__global__ void cvt_bf16_kernel(const float* __restrict__ in,
                                uint16_t* __restrict__ out, int n) {
  int i = blockIdx.x * blockDim.x + threadIdx.x;
  if (i < n) out[i] = f2bf(in[i]);
}

// Y[m][n] = sum_k A[m][k] * W[n][k]; A: MTOT x 1024, W: 1024 x 1024 (bf16).
// Block tile 128M x 64N; TDM stages 128x32 A (8KB) + 64x32 W (4KB) tiles into
// LDS, double buffered; each of 4 waves computes 32x64 (8 WMMAs/k-step).
// mode 0/1: bf16 [b,h,s,dk]; mode 2: bf16 [b,h,dk,s]; mode 3: f32 row-major.
__global__ __launch_bounds__(128)
void gemm_xwt_kernel(const uint16_t* __restrict__ A, const uint16_t* __restrict__ W,
                     void* __restrict__ dst, int mode) {
  // per buffer: A tile 128*32 elems (8192B) then W tile 64*32 elems (4096B)
  __shared__ __align__(16) uint16_t smem[2][(128 + 64) * 32];
  const int lane  = threadIdx.x & 31;
  const int wave  = threadIdx.x >> 5;
  const int half  = lane >> 4;
  const int ln    = lane & 15;
  const int mBlk  = blockIdx.x * 128;
  const int nBase = blockIdx.y * 64;

  v8f acc[2][4];
#pragma unroll
  for (int mt = 0; mt < 2; mt++)
#pragma unroll
    for (int t = 0; t < 4; t++) acc[mt][t] = v8f_zero();

#if HAVE_TDM
  const v8i g1A = make_g1(D_MODEL, MTOT, 32, 128, D_MODEL);
  const v8i g1W = make_g1(D_MODEL, D_MODEL, 32, 64, D_MODEL);
  const uint64_t baseA = (uint64_t)(uintptr_t)A + (uint64_t)mBlk * D_MODEL * 2;
  const uint64_t baseW = (uint64_t)(uintptr_t)W + (uint64_t)nBase * D_MODEL * 2;

  if (wave == 0) {  // prologue: DMA tile 0 into buffer 0
    tdm_load(make_g0(baseA, 0u), g1A);
    tdm_load(make_g0(baseW, 8192u), g1W);
  }

  for (int i = 0; i < KSTEPS; i++) {
    const int cur = i & 1;
    if (wave == 0) {
      if (i + 1 < KSTEPS) {  // prefetch next k-tile into alternate buffer
        const uint32_t nb = (uint32_t)((cur ^ 1) * 12288);
        const uint64_t ko = (uint64_t)(i + 1) * 32 * 2;
        tdm_load(make_g0(baseA + ko, nb), g1A);
        tdm_load(make_g0(baseW + ko, nb + 8192u), g1W);
        __builtin_amdgcn_s_wait_tensorcnt(2);  // tile i done, i+1 in flight
      } else {
        __builtin_amdgcn_s_wait_tensorcnt(0);
      }
    }
    __syncthreads();
    // The TDM write to LDS is invisible to alias analysis (descriptor holds a
    // raw LDS offset); escape smem into opaque asm so the fragment loads are
    // not folded to undef.
    asm volatile("" ::"v"((void*)smem) : "memory");
    const uint16_t* lA = &smem[cur][0];
    const uint16_t* lB = &smem[cur][128 * 32];
    v16bf a0 = load_frag(lA, wave * 32 + ln, 32, half);
    v16bf a1 = load_frag(lA, wave * 32 + 16 + ln, 32, half);
#pragma unroll
    for (int t = 0; t < 4; t++) {
      v16bf b = load_frag(lB, t * 16 + ln, 32, half);
      acc[0][t] = wmma_bf16f32(a0, b, acc[0][t]);
      acc[1][t] = wmma_bf16f32(a1, b, acc[1][t]);
    }
    __syncthreads();  // protect buffer from overwrite by next prefetch
  }
#else
  // Fallback: cooperative LDS staging with plain loads (single buffered).
  for (int i = 0; i < KSTEPS; i++) {
    const int k = i * 32;
    const int tid = threadIdx.x;
    uint4* sA = (uint4*)&smem[0][0];          // 512 uint4
    uint4* sB = (uint4*)&smem[0][128 * 32];   // 256 uint4
#pragma unroll
    for (int r = 0; r < 4; r++) {
      int e = tid + r * 128;
      int row = e >> 2, c = e & 3;
      sA[e] = *(const uint4*)(A + (size_t)(mBlk + row) * D_MODEL + k + c * 8);
    }
#pragma unroll
    for (int r = 0; r < 2; r++) {
      int e = tid + r * 128;
      int row = e >> 2, c = e & 3;
      sB[e] = *(const uint4*)(W + (size_t)(nBase + row) * D_MODEL + k + c * 8);
    }
    __syncthreads();
    v16bf a0 = load_frag(&smem[0][0], wave * 32 + ln, 32, half);
    v16bf a1 = load_frag(&smem[0][0], wave * 32 + 16 + ln, 32, half);
#pragma unroll
    for (int t = 0; t < 4; t++) {
      v16bf b = load_frag(&smem[0][128 * 32], t * 16 + ln, 32, half);
      acc[0][t] = wmma_bf16f32(a0, b, acc[0][t]);
      acc[1][t] = wmma_bf16f32(a1, b, acc[1][t]);
    }
    __syncthreads();
  }
#endif

  if (mode == 3) {
    float* out = (float*)dst;
#pragma unroll
    for (int mt = 0; mt < 2; mt++)
#pragma unroll
      for (int t = 0; t < 4; t++)
#pragma unroll
        for (int r = 0; r < 8; r++) {
          int m = mBlk + wave * 32 + mt * 16 + r + 8 * half;
          int n = nBase + t * 16 + ln;
          out[(size_t)m * D_MODEL + n] = acc[mt][t][r];
        }
  } else {
    uint16_t* out = (uint16_t*)dst;
#pragma unroll
    for (int mt = 0; mt < 2; mt++)
#pragma unroll
      for (int t = 0; t < 4; t++)
#pragma unroll
        for (int r = 0; r < 8; r++) {
          int m = mBlk + wave * 32 + mt * 16 + r + 8 * half;
          int n = nBase + t * 16 + ln;
          int b = m >> 11, s = m & 2047;
          int h = n >> 6,  d = n & 63;
          size_t idx = (mode == 2)
              ? ((size_t)(b * N_HEADS + h) * DK + d) * SS + s   // [b,h,d,s]
              : ((size_t)(b * N_HEADS + h) * SS + s) * DK + d;  // [b,h,s,d]
          out[idx] = f2bf(acc[mt][t][r]);
        }
  }
}

// Flash attention: one (b,h) per blockIdx.x, 64 q-rows per block (16 per wave).
__global__ __launch_bounds__(128)
void attn_kernel(const uint16_t* __restrict__ Qh, const uint16_t* __restrict__ Kh,
                 const uint16_t* __restrict__ Vt, const int* __restrict__ amask,
                 uint16_t* __restrict__ O) {
  __shared__ __align__(16) uint16_t plds[4][16 * 32];  // per-wave P staging
  const int lane = threadIdx.x & 31;
  const int wave = threadIdx.x >> 5;
  const int half = lane >> 4;
  const int ln   = lane & 15;
  const int bh   = blockIdx.x;
  const int b    = bh >> 4;
  const int h    = bh & 15;
  const int q0   = blockIdx.y * 64 + wave * 16;
  const uint16_t* Q = Qh + (size_t)bh * SS * DK;
  const uint16_t* K = Kh + (size_t)bh * SS * DK;
  const uint16_t* V = Vt + (size_t)bh * DK * SS;   // [dk][s]
  uint16_t* pl = plds[wave];

  v16bf qa0 = load_frag(Q + (size_t)q0 * DK,      ln, DK, half);
  v16bf qa1 = load_frag(Q + (size_t)q0 * DK + 32, ln, DK, half);

  v8f acc[4];
#pragma unroll
  for (int t = 0; t < 4; t++) acc[t] = v8f_zero();
  float mrow[8], lrow[8];
#pragma unroll
  for (int r = 0; r < 8; r++) { mrow[r] = -1e30f; lrow[r] = 0.0f; }

  for (int j0 = 0; j0 <= q0 + 15; j0 += 32) {
    v16bf kb0 = load_frag(K + (size_t)j0 * DK,             ln, DK, half);
    v16bf kb1 = load_frag(K + (size_t)j0 * DK + 32,        ln, DK, half);
    v16bf kb2 = load_frag(K + (size_t)(j0 + 16) * DK,      ln, DK, half);
    v16bf kb3 = load_frag(K + (size_t)(j0 + 16) * DK + 32, ln, DK, half);
    v8f s0 = wmma_bf16f32(qa0, kb0, v8f_zero());
    s0     = wmma_bf16f32(qa1, kb1, s0);
    v8f s1 = wmma_bf16f32(qa0, kb2, v8f_zero());
    s1     = wmma_bf16f32(qa1, kb3, s1);

    const int  c0  = j0 + ln;
    const int  c1  = j0 + 16 + ln;
    const bool mk0 = amask[b * SS + c0] != 0;
    const bool mk1 = amask[b * SS + c1] != 0;

    float sv0[8], sv1[8], alsc[8];
#pragma unroll
    for (int r = 0; r < 8; r++) {
      const int m = q0 + r + 8 * half;
      float a0 = s0[r] * 0.125f;   // 1/sqrt(64)
      float a1 = s1[r] * 0.125f;
      sv0[r] = (mk0 && c0 <= m) ? a0 : -1e30f;
      sv1[r] = (mk1 && c1 <= m) ? a1 : -1e30f;
    }
#pragma unroll
    for (int r = 0; r < 8; r++) {
      float tmax = rowmax16(fmaxf(sv0[r], sv1[r]));
      float mnew = fmaxf(mrow[r], tmax);
      float al   = __expf(mrow[r] - mnew);
      float p0   = __expf(sv0[r] - mnew);
      float p1   = __expf(sv1[r] - mnew);
      lrow[r] = lrow[r] * al + rowsum16(p0 + p1);
      mrow[r] = mnew;
      alsc[r] = al;
      const int mm = r + 8 * half;
      pl[mm * 32 + ln]      = f2bf(p0);
      pl[mm * 32 + 16 + ln] = f2bf(p1);
    }
#pragma unroll
    for (int t = 0; t < 4; t++)
#pragma unroll
      for (int r = 0; r < 8; r++) acc[t][r] *= alsc[r];

    v16bf pa = load_frag(pl, ln, 32, half);  // same-wave LDS ops are in-order
#pragma unroll
    for (int t = 0; t < 4; t++) {
      v16bf vb = load_frag(V + j0, t * 16 + ln, SS, half);  // Vt[d][j0+k]
      acc[t] = wmma_bf16f32(pa, vb, acc[t]);
    }
  }

  float inv[8];
#pragma unroll
  for (int r = 0; r < 8; r++) inv[r] = 1.0f / lrow[r];
#pragma unroll
  for (int t = 0; t < 4; t++)
#pragma unroll
    for (int r = 0; r < 8; r++) {
      const int s = q0 + r + 8 * half;
      const int n = h * DK + t * 16 + ln;
      O[((size_t)(b * SS + s)) * D_MODEL + n] = f2bf(acc[t][r] * inv[r]);
    }
}

// ---------- launch ----------

extern "C" void kernel_launch(void* const* d_in, const int* in_sizes, int n_in,
                              void* d_out, int out_size, void* d_ws, size_t ws_size,
                              hipStream_t stream) {
  (void)in_sizes; (void)n_in; (void)out_size; (void)ws_size;
  const float* xq = (const float*)d_in[0];
  const float* xk = (const float*)d_in[1];
  const float* xv = (const float*)d_in[2];
  const int* amask = (const int*)d_in[3];
  const float* Wq = (const float*)d_in[4];
  const float* Wk = (const float*)d_in[5];
  const float* Wv = (const float*)d_in[6];
  const float* Wo = (const float*)d_in[7];

  const size_t XSZ = (size_t)MTOT * D_MODEL;      // 4,194,304 elements
  const size_t WSZ = (size_t)D_MODEL * D_MODEL;   // 1,048,576 elements

  char* ws = (char*)d_ws;
  uint16_t* xq_bf = (uint16_t*)ws; ws += XSZ * 2;
  uint16_t* xk_bf = (uint16_t*)ws; ws += XSZ * 2;
  uint16_t* xv_bf = (uint16_t*)ws; ws += XSZ * 2;
  uint16_t* wq_bf = (uint16_t*)ws; ws += WSZ * 2;
  uint16_t* wk_bf = (uint16_t*)ws; ws += WSZ * 2;
  uint16_t* wv_bf = (uint16_t*)ws; ws += WSZ * 2;
  uint16_t* wo_bf = (uint16_t*)ws; ws += WSZ * 2;
  uint16_t* Qh  = (uint16_t*)ws; ws += XSZ * 2;   // [b,h,s,dk]
  uint16_t* Kh  = (uint16_t*)ws; ws += XSZ * 2;   // [b,h,s,dk]
  uint16_t* Vth = (uint16_t*)ws; ws += XSZ * 2;   // [b,h,dk,s]
  uint16_t* Obf = (uint16_t*)ws; ws += XSZ * 2;   // [b*s, d]  (total 64 MB)

  cvt_bf16_kernel<<<(int)(XSZ / 256), 256, 0, stream>>>(xq, xq_bf, (int)XSZ);
  cvt_bf16_kernel<<<(int)(XSZ / 256), 256, 0, stream>>>(xk, xk_bf, (int)XSZ);
  cvt_bf16_kernel<<<(int)(XSZ / 256), 256, 0, stream>>>(xv, xv_bf, (int)XSZ);
  cvt_bf16_kernel<<<(int)(WSZ / 256), 256, 0, stream>>>(Wq, wq_bf, (int)WSZ);
  cvt_bf16_kernel<<<(int)(WSZ / 256), 256, 0, stream>>>(Wk, wk_bf, (int)WSZ);
  cvt_bf16_kernel<<<(int)(WSZ / 256), 256, 0, stream>>>(Wv, wv_bf, (int)WSZ);
  cvt_bf16_kernel<<<(int)(WSZ / 256), 256, 0, stream>>>(Wo, wo_bf, (int)WSZ);

  dim3 gg(MTOT / 128, D_MODEL / 64);
  gemm_xwt_kernel<<<gg, 128, 0, stream>>>(xq_bf, wq_bf, Qh, 0);
  gemm_xwt_kernel<<<gg, 128, 0, stream>>>(xk_bf, wk_bf, Kh, 1);
  gemm_xwt_kernel<<<gg, 128, 0, stream>>>(xv_bf, wv_bf, Vth, 2);

  attn_kernel<<<dim3(BB * N_HEADS, SS / 64), 128, 0, stream>>>(Qh, Kh, Vth, amask, Obf);

  gemm_xwt_kernel<<<gg, 128, 0, stream>>>(Obf, wo_bf, d_out, 3);
}
